// GlycanGNNEncoder_7069516169549
// MI455X (gfx1250) — compile-verified, matrix operand order checked by reading, not run on hardware
//
#include <hip/hip_runtime.h>
#include <hip/hip_bf16.h>

// ---------------------------------------------------------------------------
// GlycanGNNEncoder for MI455X (gfx1250, wave32, WMMA f16 16x16x32, f32 accum)
//
// N=100000 nodes, E=800000 edges, G=1024 graphs, IN=5, H=64, D=512.
// Edge gather/scatter dominates (~1.2 GB, L2-resident: h+agg = 51MB << 192MB
// L2). Dense GEMMs run on v_wmma_f32_16x16x32_f16 with fp32 accumulation.
// Data movement:
//   - fragment-linear pre-swizzled f16 weights in LDS (ds_load_b128 pairs)
//   - direct global float4 A-fragment builds for GEMM1 (no LDS staging)
//   - f16 transpose tile (stride 88 halves) for GEMM2 A (v8h LDS loads)
//   - pre-swizzled f16 proj_W fragments in workspace for k_proj
//   - B-fragment groups preloaded before each WMMA chain so LDS latency
//     overlaps matrix issue instead of stalling on s_wait_dscnt 0x0
// ---------------------------------------------------------------------------

typedef __attribute__((ext_vector_type(16))) _Float16 v16h;
typedef __attribute__((ext_vector_type(8)))  _Float16 v8h;
typedef __attribute__((ext_vector_type(8)))  float    v8f;

#define N_NODES 100000
#define N_EDGES 800000
#define N_GRAPH 1024
#define F_IN    5
#define H_DIM   64
#define D_DIM   512
#define ED_DIM  3
#define BN_EPS  1e-5f
#define LN_EPS  1e-5f
#define YSTRIDE 88   // f16 row stride for transpose tile: 176B rows keep 16B
                     // chunk alignment and distinct start banks for 16 rows

union AFrag { v16h v; v8h h2[2]; _Float16 e[16]; };
union CFrag { v8f  v; float e[8]; };

// A/B fragment element i -> K offset within the 32-wide K slab, for lane half hf.
__device__ __forceinline__ int kmap(int i, int hf) {
  return (i < 8) ? (hf * 8 + i) : (16 + hf * 8 + (i - 8));
}

// ---------------------------------------------------------------------------
// h = x @ np_W + np_b        [N,5] x [5,64]
// ---------------------------------------------------------------------------
__global__ __launch_bounds__(256)
void k_node_proj(const float* __restrict__ x, const float* __restrict__ W,
                 const float* __restrict__ b, float* __restrict__ h) {
  int gid = blockIdx.x * 256 + threadIdx.x;
  if (gid >= N_NODES * H_DIM) return;
  int n = gid >> 6, c = gid & 63;
  float acc = b[c];
  const float* xr = x + n * F_IN;
  #pragma unroll
  for (int k = 0; k < F_IN; ++k) acc = fmaf(xr[k], W[k * H_DIM + c], acc);
  h[gid] = acc;
}

// ---------------------------------------------------------------------------
// agg[dst] += relu(h[src] + edge_attr @ We + be)   (segment-sum via f32 atomics)
// ---------------------------------------------------------------------------
__global__ __launch_bounds__(256)
void k_edge(const float* __restrict__ h, const float* __restrict__ ea,
            const float* __restrict__ We, const float* __restrict__ be,
            const int* __restrict__ src, const int* __restrict__ dst,
            float* __restrict__ agg) {
  int gid = blockIdx.x * 256 + threadIdx.x;
  if (gid >= N_EDGES * H_DIM) return;
  int e = gid >> 6, c = gid & 63;
  int s = src[e], d = dst[e];
  const float* ar = ea + e * ED_DIM;
  float m = h[s * H_DIM + c] + be[c];
  m = fmaf(ar[0], We[0 * H_DIM + c], m);
  m = fmaf(ar[1], We[1 * H_DIM + c], m);
  m = fmaf(ar[2], We[2 * H_DIM + c], m);
  m = fmaxf(m, 0.0f);
  atomicAdd(agg + d * H_DIM + c, m);
}

// ---------------------------------------------------------------------------
// h = relu(BN(relu((h+agg) @ W1 + b1) @ W2 + b2))   in place, WMMA f16.
// Block = 256 threads = 8 waves; each wave owns a 16-node tile (N % 16 == 0,
// so tiles are whole; OOB waves clamp loads and predicate only final stores,
// keeping EXEC all-ones through every WMMA).
// ---------------------------------------------------------------------------
__global__ __launch_bounds__(256)
void k_mlp(float* __restrict__ h, const float* __restrict__ agg,
           const float* __restrict__ W1, const float* __restrict__ b1,
           const float* __restrict__ W2, const float* __restrict__ b2,
           const float* __restrict__ bn_g, const float* __restrict__ bn_b,
           const float* __restrict__ bn_rm, const float* __restrict__ bn_rv) {
  // Fragment-linear weight layouts: frag (kt,nt,lane) -> 16 contiguous halves.
  __shared__ alignas(32) _Float16 sW1s[2 * 4 * 32 * 16];
  __shared__ alignas(32) _Float16 sW2s[2 * 4 * 32 * 16];
  __shared__ float sB1[H_DIM], sB2[H_DIM], sSc[H_DIM], sSh[H_DIM];
  __shared__ alignas(32) _Float16 sY[8 * 16 * YSTRIDE];  // per-wave y1 tiles

  const int tid  = threadIdx.x;
  const int w    = tid >> 5, lane = tid & 31;
  const int hf   = lane >> 4, r = lane & 15;

  // Stage weights pre-swizzled: thread tid owns fragment tid (one per thread).
  {
    const int skt = tid >> 7;           // 0..1
    const int snt = (tid >> 5) & 3;     // 0..3
    const int sln = tid & 31;
    const int shf = sln >> 4, srr = sln & 15;
    const int col = snt * 16 + srr;
    #pragma unroll
    for (int i = 0; i < 16; ++i) {
      const int k = skt * 32 + kmap(i, shf);
      sW1s[tid * 16 + i] = (_Float16)W1[k * H_DIM + col];
      sW2s[tid * 16 + i] = (_Float16)W2[k * H_DIM + col];
    }
  }
  if (tid < H_DIM) {
    sB1[tid] = b1[tid];
    sB2[tid] = b2[tid];
    float sc = bn_g[tid] * rsqrtf(bn_rv[tid] + BN_EPS);
    sSc[tid] = sc;
    sSh[tid] = bn_b[tid] - bn_rm[tid] * sc;
  }
  __syncthreads();

  const int base = blockIdx.x * 128 + w * 16;
  _Float16* yt = sY + w * 16 * YSTRIDE;

  // Build GEMM1 A-fragments straight from global: lane (hf,r) feeds row r with
  // K chunks {8hf, 16+8hf, 32+8hf, 48+8hf}..+7 -> aligned float4 loads.
  AFrag a1[2];
  {
    const int n   = base + r;
    const int nc  = (n < N_NODES) ? n : (N_NODES - 1);
    const float4* hp = (const float4*)(h + (size_t)nc * H_DIM);
    const float4* ap = (const float4*)(agg + (size_t)nc * H_DIM);
    #pragma unroll
    for (int j = 0; j < 4; ++j) {              // chunk j: cols 16*j + 8*hf .. +7
      const int fi = 2 * hf + 4 * j;           // float4 index within the row
      float4 h0 = hp[fi], h1 = hp[fi + 1];
      float4 g0 = ap[fi], g1 = ap[fi + 1];
      _Float16* dp = &a1[j >> 1].e[(j & 1) * 8];
      dp[0] = (_Float16)(h0.x + g0.x);
      dp[1] = (_Float16)(h0.y + g0.y);
      dp[2] = (_Float16)(h0.z + g0.z);
      dp[3] = (_Float16)(h0.w + g0.w);
      dp[4] = (_Float16)(h1.x + g1.x);
      dp[5] = (_Float16)(h1.y + g1.y);
      dp[6] = (_Float16)(h1.z + g1.z);
      dp[7] = (_Float16)(h1.w + g1.w);
    }
  }

  // GEMM1: y1 = z @ W1. Preload each kt-group of 4 B fragments so the
  // ds_load_b128 clause overlaps the WMMA chain instead of stalling per-op.
  CFrag acc[4];
  #pragma unroll
  for (int nt = 0; nt < 4; ++nt)
    #pragma unroll
    for (int i = 0; i < 8; ++i) acc[nt].e[i] = 0.0f;

  #pragma unroll
  for (int kt = 0; kt < 2; ++kt) {
    v16h bf[4];
    #pragma unroll
    for (int nt = 0; nt < 4; ++nt)
      bf[nt] = *(const v16h*)&sW1s[((kt * 4 + nt) * 32 + lane) * 16];
    #pragma unroll
    for (int nt = 0; nt < 4; ++nt)
      acc[nt].v = __builtin_amdgcn_wmma_f32_16x16x32_f16(
          false, a1[kt].v, false, bf[nt], (short)0, acc[nt].v, false, false);
  }

  // relu(y1 + b1) -> f16 LDS tile (C/D layout: M = vgpr + 8*hf, N = r)
  #pragma unroll
  for (int nt = 0; nt < 4; ++nt) {
    const int col = nt * 16 + r;
    #pragma unroll
    for (int v = 0; v < 8; ++v)
      yt[(v + 8 * hf) * YSTRIDE + col] =
          (_Float16)fmaxf(acc[nt].e[v] + sB1[col], 0.0f);
  }
  __syncthreads();

  // GEMM2: y2 = y1 @ W2   (A fragments: two aligned 16B LDS vector loads)
  #pragma unroll
  for (int nt = 0; nt < 4; ++nt)
    #pragma unroll
    for (int i = 0; i < 8; ++i) acc[nt].e[i] = 0.0f;

  #pragma unroll
  for (int kt = 0; kt < 2; ++kt) {
    AFrag a;
    const int k0 = kt * 32;
    a.h2[0] = *(const v8h*)&yt[r * YSTRIDE + k0 + 8 * hf];
    a.h2[1] = *(const v8h*)&yt[r * YSTRIDE + k0 + 16 + 8 * hf];
    v16h bf[4];
    #pragma unroll
    for (int nt = 0; nt < 4; ++nt)
      bf[nt] = *(const v16h*)&sW2s[((kt * 4 + nt) * 32 + lane) * 16];
    #pragma unroll
    for (int nt = 0; nt < 4; ++nt)
      acc[nt].v = __builtin_amdgcn_wmma_f32_16x16x32_f16(
          false, a.v, false, bf[nt], (short)0, acc[nt].v, false, false);
  }

  // Epilogue: BN (fused scale/shift) + relu, store h in place
  #pragma unroll
  for (int nt = 0; nt < 4; ++nt) {
    const int col = nt * 16 + r;
    #pragma unroll
    for (int v = 0; v < 8; ++v) {
      const int n = base + v + 8 * hf;
      if (n < N_NODES) {
        float y = acc[nt].e[v] + sB2[col];
        h[(size_t)n * H_DIM + col] = fmaxf(fmaf(y, sSc[col], sSh[col]), 0.0f);
      }
    }
  }
}

// ---------------------------------------------------------------------------
// Pooling: per-graph sum / count / max. h >= 0 post-relu, so uint atomicMax on
// the bit pattern with zero-init matches segment_max + empty-segment guard.
// ---------------------------------------------------------------------------
__global__ __launch_bounds__(256)
void k_pool(const float* __restrict__ h, const int* __restrict__ batch,
            float* __restrict__ gsum, float* __restrict__ gmaxb,
            float* __restrict__ gcnt) {
  int gid = blockIdx.x * 256 + threadIdx.x;
  if (gid >= N_NODES * H_DIM) return;
  int n = gid >> 6, c = gid & 63;
  int b = batch[n];
  float v = h[gid];
  atomicAdd(gsum + b * H_DIM + c, v);
  atomicMax((unsigned int*)(gmaxb + b * H_DIM + c), __float_as_uint(v));
  if (c == 0) atomicAdd(gcnt + b, 1.0f);
}

__global__ __launch_bounds__(256)
void k_build_g(const float* __restrict__ gsum, const float* __restrict__ gmaxb,
               const float* __restrict__ gcnt, _Float16* __restrict__ g16) {
  int gid = blockIdx.x * 256 + threadIdx.x;
  if (gid >= N_GRAPH * 128) return;
  int g = gid >> 7, c = gid & 127;
  float v = (c < H_DIM) ? gsum[g * H_DIM + c] / fmaxf(gcnt[g], 1.0f)
                        : gmaxb[g * H_DIM + (c - H_DIM)];
  g16[gid] = (_Float16)v;
}

// ---------------------------------------------------------------------------
// Pre-swizzle proj_W [128,512] into f16 fragment-linear layout:
// frag (kt in 0..3, tn in 0..31, lane in 0..31) -> 16 contiguous halves.
// ---------------------------------------------------------------------------
__global__ __launch_bounds__(256)
void k_swz_proj(const float* __restrict__ Wp, _Float16* __restrict__ Wp16) {
  int tid = blockIdx.x * 256 + threadIdx.x;          // 4096 fragments
  if (tid >= 4 * 32 * 32) return;
  const int kt = tid >> 10;
  const int ln = tid & 31;
  const int hf = ln >> 4, rr = ln & 15;
  const int tn = (tid >> 5) & 31;
  const int col = tn * 16 + rr;
  #pragma unroll
  for (int i = 0; i < 16; ++i) {
    const int k = kt * 32 + kmap(i, hf);
    Wp16[tid * 16 + i] = (_Float16)Wp[k * D_DIM + col];
  }
}

// ---------------------------------------------------------------------------
// out = g @ proj_W + proj_b    [1024,128] x [128,512], one 16x16 tile per wave
// All A and B fragments preloaded so global load latency overlaps the chain.
// ---------------------------------------------------------------------------
__global__ __launch_bounds__(256)
void k_proj(const _Float16* __restrict__ g16, const _Float16* __restrict__ Wp16,
            const float* __restrict__ bp, float* __restrict__ out) {
  const int w = threadIdx.x >> 5, lane = threadIdx.x & 31;
  const int hf = lane >> 4, r = lane & 15;
  const int tile = blockIdx.x * 8 + w;      // 2048 tiles = (1024/16)*(512/16)
  const int tm = tile >> 5;
  const int tn = tile & 31;

  CFrag acc;
  #pragma unroll
  for (int i = 0; i < 8; ++i) acc.e[i] = 0.0f;

  const int rowA = tm * 16 + r;
  AFrag a[4];
  v16h bf[4];
  #pragma unroll
  for (int kt = 0; kt < 4; ++kt) {
    const int k0 = kt * 32;
    a[kt].h2[0] = *(const v8h*)&g16[rowA * 128 + k0 + 8 * hf];
    a[kt].h2[1] = *(const v8h*)&g16[rowA * 128 + k0 + 16 + 8 * hf];
    bf[kt] = *(const v16h*)&Wp16[((kt * 32 + tn) * 32 + lane) * 16];
  }
  #pragma unroll
  for (int kt = 0; kt < 4; ++kt)
    acc.v = __builtin_amdgcn_wmma_f32_16x16x32_f16(
        false, a[kt].v, false, bf[kt], (short)0, acc.v, false, false);

  const int col = tn * 16 + r;
  #pragma unroll
  for (int v = 0; v < 8; ++v)
    out[(tm * 16 + v + 8 * hf) * D_DIM + col] = acc.e[v] + bp[col];
}

// ---------------------------------------------------------------------------
// In-place LayerNorm + relu over D=512 per graph row. Block = 256 threads.
// ---------------------------------------------------------------------------
__global__ __launch_bounds__(256)
void k_ln(float* __restrict__ out, const float* __restrict__ ln_g,
          const float* __restrict__ ln_b) {
  __shared__ float s1[256], s2[256];
  const int g = blockIdx.x, tid = threadIdx.x;
  float* row = out + g * D_DIM;
  float v0 = row[tid], v1 = row[tid + 256];
  s1[tid] = v0 + v1;
  s2[tid] = v0 * v0 + v1 * v1;
  __syncthreads();
  for (int s = 128; s > 0; s >>= 1) {
    if (tid < s) { s1[tid] += s1[tid + s]; s2[tid] += s2[tid + s]; }
    __syncthreads();
  }
  const float mean = s1[0] * (1.0f / D_DIM);
  const float var  = s2[0] * (1.0f / D_DIM) - mean * mean;
  const float inv  = rsqrtf(var + LN_EPS);
  row[tid]       = fmaxf((v0 - mean) * inv * ln_g[tid]       + ln_b[tid],       0.0f);
  row[tid + 256] = fmaxf((v1 - mean) * inv * ln_g[tid + 256] + ln_b[tid + 256], 0.0f);
}

// ---------------------------------------------------------------------------
// Host orchestration. Input flattening assumption (JAX sorts dict keys inside
// the params pytree; top level keeps setup_inputs() insertion order):
//   0:x  1:edge_attr
//   2..31: layers l=0..2, each 10 leaves in sorted order:
//          W1, W2, We, b1, b2, be, bn_b, bn_g, bn_rm, bn_rv
//   32:ln_b 33:ln_g 34:np_W 35:np_b 36:proj_W 37:proj_b
//   38:edge_index[2,E] 39:batch[N]
// ---------------------------------------------------------------------------
extern "C" void kernel_launch(void* const* d_in, const int* in_sizes, int n_in,
                              void* d_out, int out_size, void* d_ws, size_t ws_size,
                              hipStream_t stream) {
  (void)in_sizes; (void)n_in; (void)out_size; (void)ws_size;

  const float* x  = (const float*)d_in[0];
  const float* ea = (const float*)d_in[1];

  const float *W1[3], *W2[3], *We[3], *b1[3], *b2[3], *be[3];
  const float *bn_b[3], *bn_g[3], *bn_rm[3], *bn_rv[3];
  for (int l = 0; l < 3; ++l) {
    int base = 2 + 10 * l;
    W1[l]    = (const float*)d_in[base + 0];
    W2[l]    = (const float*)d_in[base + 1];
    We[l]    = (const float*)d_in[base + 2];
    b1[l]    = (const float*)d_in[base + 3];
    b2[l]    = (const float*)d_in[base + 4];
    be[l]    = (const float*)d_in[base + 5];
    bn_b[l]  = (const float*)d_in[base + 6];
    bn_g[l]  = (const float*)d_in[base + 7];
    bn_rm[l] = (const float*)d_in[base + 8];
    bn_rv[l] = (const float*)d_in[base + 9];
  }
  const float* ln_b   = (const float*)d_in[32];
  const float* ln_g   = (const float*)d_in[33];
  const float* np_W   = (const float*)d_in[34];
  const float* np_b   = (const float*)d_in[35];
  const float* proj_W = (const float*)d_in[36];
  const float* proj_b = (const float*)d_in[37];
  const int* edge_index = (const int*)d_in[38];
  const int* batch      = (const int*)d_in[39];
  const int* src = edge_index;
  const int* dst = edge_index + N_EDGES;

  // Workspace carve-up (~52.1 MB)
  char* ws = (char*)d_ws;
  float* h     = (float*)ws; ws += (size_t)N_NODES * H_DIM * sizeof(float);
  float* agg   = (float*)ws; ws += (size_t)N_NODES * H_DIM * sizeof(float);
  float* gsum  = (float*)ws; ws += (size_t)N_GRAPH * H_DIM * sizeof(float);
  float* gmaxb = (float*)ws; ws += (size_t)N_GRAPH * H_DIM * sizeof(float);
  float* gcnt  = (float*)ws; ws += (size_t)N_GRAPH * sizeof(float);
  _Float16* g16  = (_Float16*)ws; ws += (size_t)N_GRAPH * 128 * sizeof(_Float16);
  _Float16* Wp16 = (_Float16*)ws;

  float* out = (float*)d_out;

  // Static-weight pre-swizzle can run up front, overlapping nothing it needs.
  k_swz_proj<<<16, 256, 0, stream>>>(proj_W, Wp16);

  k_node_proj<<<(N_NODES * H_DIM + 255) / 256, 256, 0, stream>>>(x, np_W, np_b, h);

  for (int l = 0; l < 3; ++l) {
    hipMemsetAsync(agg, 0, (size_t)N_NODES * H_DIM * sizeof(float), stream);
    k_edge<<<(N_EDGES * H_DIM + 255) / 256, 256, 0, stream>>>(
        h, ea, We[l], be[l], src, dst, agg);
    k_mlp<<<(N_NODES + 127) / 128, 256, 0, stream>>>(
        h, agg, W1[l], b1[l], W2[l], b2[l], bn_g[l], bn_b[l], bn_rm[l], bn_rv[l]);
  }

  hipMemsetAsync(gsum,  0, (size_t)N_GRAPH * H_DIM * sizeof(float), stream);
  hipMemsetAsync(gmaxb, 0, (size_t)N_GRAPH * H_DIM * sizeof(float), stream);
  hipMemsetAsync(gcnt,  0, (size_t)N_GRAPH * sizeof(float), stream);
  k_pool<<<(N_NODES * H_DIM + 255) / 256, 256, 0, stream>>>(h, batch, gsum, gmaxb, gcnt);
  k_build_g<<<(N_GRAPH * 128 + 255) / 256, 256, 0, stream>>>(gsum, gmaxb, gcnt, g16);

  k_proj<<<(N_GRAPH / 16) * (D_DIM / 16) / 8, 256, 0, stream>>>(g16, Wp16, proj_b, out);
  k_ln<<<N_GRAPH, 256, 0, stream>>>(out, ln_g, ln_b);
}
